// CamouflageDetectionNet_10969346474343
// MI455X (gfx1250) — matrix operand order repaired and checked by reading, compiled, not run on hardware
//
#include <hip/hip_runtime.h>
#include <cstdint>
#include <cstddef>

#define EPS_ 1e-5f

typedef __attribute__((ext_vector_type(16))) __bf16 v16bf;
typedef __attribute__((ext_vector_type(8)))  float  v8f;

__device__ __forceinline__ float sigmoidf_(float x){ return 1.0f/(1.0f+__expf(-x)); }
__device__ __forceinline__ float siluf_(float x){ return x*sigmoidf_(x); }
__device__ __forceinline__ unsigned short f2bf_(float f){
  unsigned int u = __float_as_uint(f);
  u += 0x7FFFu + ((u>>16)&1u);
  return (unsigned short)(u>>16);
}

// ---------------------------------------------------------------------------
// bf16 staging
// ---------------------------------------------------------------------------
__global__ void k_f32_to_bf16(const float* __restrict__ src, unsigned short* __restrict__ dst, int n){
  int i = blockIdx.x*blockDim.x + threadIdx.x;
  if (i < n) dst[i] = f2bf_(src[i]);
}

// Stage weight W (N rows, K cols, f32) into B matrix (K, Npad) bf16 with
// zero-filled pad columns: dst[k*Npad + n] = W[n*K + k]  (n < N), else 0.
__global__ void k_stageB(const float* __restrict__ src, unsigned short* __restrict__ dst,
                         int N, int K, int Npad){
  int i = blockIdx.x*blockDim.x + threadIdx.x;
  if (i >= K*Npad) return;
  int k = i / Npad, n = i % Npad;
  dst[i] = (n < N) ? f2bf_(src[(size_t)n*K + k]) : (unsigned short)0;
}

// ---------------------------------------------------------------------------
// WMMA bf16 GEMM: C[M,N] = A[M,K] * B[K,Npad]  (A,B bf16 staged, C f32)
// one wave (32 lanes) per 16x32 output tile (2 accumulators, A reused);
// K multiple of 32, Npad multiple of 32, M multiple of 16. Branch-free loop.
// ---------------------------------------------------------------------------
__global__ void k_gemm_bf16_wmma(const unsigned short* __restrict__ A,
                                 const unsigned short* __restrict__ B,
                                 float* __restrict__ C, int M, int N, int K, int Npad){
  const int n0 = blockIdx.x * 32;
  const int m0 = blockIdx.y * 16;
  const int lane = threadIdx.x;           // 0..31
  v8f acc0 = {0.f,0.f,0.f,0.f,0.f,0.f,0.f,0.f};
  v8f acc1 = acc0;

  const int am = m0 + (lane & 15);
  const int kbase = (lane >> 4) * 8;

  const unsigned short* arow = A + (size_t)am * K + kbase;
  const unsigned short* brow = B + (size_t)kbase * Npad + n0 + (lane & 15);

  for (int k0 = 0; k0 < K; k0 += 32) {
    union { v16bf v; unsigned short u[16]; } a, b0, b1;
#pragma unroll
    for (int e = 0; e < 8; ++e) { a.u[e] = arow[e]; a.u[8+e] = arow[16+e]; }
#pragma unroll
    for (int e = 0; e < 8; ++e) { b0.u[e] = brow[(size_t)e*Npad]; b0.u[8+e] = brow[(size_t)(16+e)*Npad]; b1.u[e] = brow[(size_t)e*Npad + 16]; b1.u[8+e] = brow[(size_t)(16+e)*Npad + 16]; }
    acc0 = __builtin_amdgcn_wmma_f32_16x16x32_bf16(false, a.v, false, b0.v, (short)0, acc0, false, false);
    acc1 = __builtin_amdgcn_wmma_f32_16x16x32_bf16(false, a.v, false, b1.v, (short)0, acc1, false, false);
    arow += 32;
    brow += (size_t)32 * Npad;
  }

  const int cn0 = n0 + (lane & 15);
  const int cn1 = cn0 + 16;
  const int mrow0 = m0 + ((lane >> 4) ? 8 : 0);
  if (cn0 < N) {
#pragma unroll
    for (int r = 0; r < 8; ++r) C[(size_t)(mrow0 + r) * N + cn0] = acc0[r];
  }
  if (cn1 < N) {
#pragma unroll
    for (int r = 0; r < 8; ++r) C[(size_t)(mrow0 + r) * N + cn1] = acc1[r];
  }
}

// ---------------------------------------------------------------------------
// Convolutions (direct, f32 — bandwidth-trivial at these sizes)
// ---------------------------------------------------------------------------
__global__ void k_conv3x3(const float* __restrict__ x, const float* __restrict__ w,
                          float* __restrict__ y, int B, int Cin, int Cout, int H, int W){
  int idx = blockIdx.x*blockDim.x + threadIdx.x;
  int total = B*Cout*H*W;
  if (idx >= total) return;
  int wc = idx % W, h = (idx/W) % H, co = (idx/(W*H)) % Cout, b = idx/(W*H*Cout);
  float acc = 0.f;
  for (int ci = 0; ci < Cin; ++ci) {
    const float* xc = x + ((size_t)(b*Cin+ci))*H*W;
    const float* wk = w + ((size_t)(co*Cin+ci))*9;
#pragma unroll
    for (int kh = -1; kh <= 1; ++kh) {
      int hh = h + kh;
      if (hh < 0 || hh >= H) continue;
#pragma unroll
      for (int kw = -1; kw <= 1; ++kw) {
        int ww = wc + kw;
        if (ww < 0 || ww >= W) continue;
        acc += xc[hh*W + ww] * wk[(kh+1)*3 + (kw+1)];
      }
    }
  }
  y[idx] = acc;
}

__global__ void k_conv1x1(const float* __restrict__ x, const float* __restrict__ w,
                          const float* __restrict__ bias, float* __restrict__ y,
                          int B, int Cin, int Cout, int H, int W){
  int idx = blockIdx.x*blockDim.x + threadIdx.x;
  int HW = H*W, total = B*Cout*HW;
  if (idx >= total) return;
  int i = idx % HW, co = (idx/HW) % Cout, b = idx/(HW*Cout);
  float acc = bias ? bias[co] : 0.f;
  for (int ci = 0; ci < Cin; ++ci)
    acc += x[((size_t)(b*Cin+ci))*HW + i] * w[co*Cin + ci];
  y[idx] = acc;
}

__global__ void k_conv7x7_2to1(const float* __restrict__ cp, const float* __restrict__ w,
                               float* __restrict__ s, int B, int H, int W){
  int idx = blockIdx.x*blockDim.x + threadIdx.x;
  int HW = H*W, total = B*HW;
  if (idx >= total) return;
  int i = idx % HW, b = idx/HW;
  int h = i / W, wc = i % W;
  float acc = 0.f;
  for (int ci = 0; ci < 2; ++ci) {
    const float* xc = cp + ((size_t)(b*2+ci))*HW;
    const float* wk = w + ci*49;
    for (int kh = -3; kh <= 3; ++kh) {
      int hh = h + kh; if (hh < 0 || hh >= H) continue;
      for (int kw = -3; kw <= 3; ++kw) {
        int ww = wc + kw; if (ww < 0 || ww >= W) continue;
        acc += xc[hh*W + ww] * wk[(kh+3)*7 + (kw+3)];
      }
    }
  }
  s[idx] = acc;
}

// ---------------------------------------------------------------------------
// BatchNorm (batch statistics over (B,H,W) per channel)
// ---------------------------------------------------------------------------
__global__ void k_bn_stats(const float* __restrict__ x, float* __restrict__ mean,
                           float* __restrict__ var, int C, int HW, int B){
  __shared__ float s_sum[256], s_sq[256];
  int c = blockIdx.x, tid = threadIdx.x;
  float sum = 0.f, sq = 0.f;
  for (int t = tid; t < B*HW; t += 256) {
    int b = t / HW, i = t % HW;
    float v = x[((size_t)b*C + c)*HW + i];
    sum += v; sq += v*v;
  }
  s_sum[tid] = sum; s_sq[tid] = sq;
  __syncthreads();
  for (int st = 128; st > 0; st >>= 1) {
    if (tid < st) { s_sum[tid] += s_sum[tid+st]; s_sq[tid] += s_sq[tid+st]; }
    __syncthreads();
  }
  if (tid == 0) {
    float n = (float)(B*HW);
    float m = s_sum[0]/n;
    mean[c] = m;
    var[c] = s_sq[0]/n - m*m;
  }
}

__global__ void k_bn_apply(const float* __restrict__ x, float* __restrict__ y,
                           const float* __restrict__ mean, const float* __restrict__ var,
                           const float* __restrict__ g, const float* __restrict__ b,
                           int C, int HW, int total, int relu){
  int idx = blockIdx.x*blockDim.x + threadIdx.x;
  if (idx >= total) return;
  int c = (idx / HW) % C;
  float v = (x[idx] - mean[c]) * rsqrtf(var[c] + EPS_) * g[c] + b[c];
  if (relu) v = fmaxf(v, 0.f);
  y[idx] = v;
}

// ---------------------------------------------------------------------------
// Pool to 16x16 and emit sequence layout (B, 256, C)
// ---------------------------------------------------------------------------
__global__ void k_pool_seq(const float* __restrict__ h, float* __restrict__ seq,
                           int B, int C, int H, int W){
  int idx = blockIdx.x*blockDim.x + threadIdx.x;
  int total = B*C*256;
  if (idx >= total) return;
  int l = idx % 256, c = (idx/256) % C, b = idx/(256*C);
  int i = l / 16, j = l % 16;
  int k = H / 16;
  float s = 0.f;
  const float* base = h + ((size_t)(b*C + c))*H*W;
  for (int a = 0; a < k; ++a)
    for (int d = 0; d < k; ++d)
      s += base[(i*k + a)*W + (j*k + d)];
  seq[((size_t)(b*256 + l))*C + c] = s / (float)(k*k);
}

// ---------------------------------------------------------------------------
// Mamba: causal conv1d (kernel 4) + SiLU
// ---------------------------------------------------------------------------
__global__ void k_conv1d_silu(const float* __restrict__ xz, const float* __restrict__ cw,
                              const float* __restrict__ cb, float* __restrict__ xc,
                              int B, int L, int di){
  int idx = blockIdx.x*blockDim.x + threadIdx.x;
  int total = B*L*di;
  if (idx >= total) return;
  int d = idx % di, l = (idx/di) % L, b = idx/(di*L);
  float acc = cb[d];
#pragma unroll
  for (int k = 0; k < 4; ++k) {
    int ls = l + k - 3;
    if (ls >= 0)
      acc += xz[((size_t)(b*L + ls))*(2*di) + d] * cw[d*4 + k];
  }
  xc[idx] = siluf_(acc);
}

// dt = softplus(dtl @ dt_w.T + dt_b); dtl = dbl[:, :dr]
__global__ void k_dtproj(const float* __restrict__ dbl, const float* __restrict__ dt_w,
                         const float* __restrict__ dt_b, float* __restrict__ dtv,
                         int M, int di, int dr, int NC){
  int idx = blockIdx.x*blockDim.x + threadIdx.x;
  if (idx >= M*di) return;
  int d = idx % di, m = idx / di;
  float acc = dt_b[d];
  const float* row = dbl + (size_t)m*NC;
  for (int j = 0; j < dr; ++j) acc += row[j] * dt_w[d*dr + j];
  dtv[idx] = (acc > 20.f) ? acc : log1pf(__expf(acc));
}

// Selective scan: one thread per (b,d); h[64], a[64] register-resident.
__global__ void k_scan(const float* __restrict__ dtv, const float* __restrict__ xc,
                       const float* __restrict__ dbl, const float* __restrict__ xz,
                       const float* __restrict__ A_log, const float* __restrict__ Dp,
                       float* __restrict__ y, int B, int di, int dr, int NC){
  int idx = blockIdx.x*blockDim.x + threadIdx.x;
  if (idx >= B*di) return;
  int d = idx % di, b = idx / di;
  float a[64], h[64];
#pragma unroll
  for (int n = 0; n < 64; ++n) { a[n] = -__expf(A_log[(size_t)d*64 + n]); h[n] = 0.f; }
  const float Dd = Dp[d];
  for (int l = 0; l < 256; ++l) {
    size_t m = (size_t)b*256 + l;
    float dt = dtv[m*di + d];
    float xv = xc[m*di + d];
    const float* Brow = dbl + m*NC + dr;
    const float* Crow = Brow + 64;
    float acc = 0.f;
#pragma unroll
    for (int n = 0; n < 64; ++n) {
      h[n] = __expf(dt * a[n]) * h[n] + dt * Brow[n] * xv;
      acc += h[n] * Crow[n];
    }
    float zv = xz[m*(size_t)(2*di) + di + d];
    y[m*di + d] = (acc + Dd * xv) * siluf_(zv);
  }
}

// ---------------------------------------------------------------------------
// bilinear 16x16 -> (H,W) from sequence layout, + identity, relu
// ---------------------------------------------------------------------------
__global__ void k_resize_add_relu(const float* __restrict__ mseq, const float* __restrict__ ident,
                                  float* __restrict__ out, int B, int C, int H, int W){
  int idx = blockIdx.x*blockDim.x + threadIdx.x;
  int total = B*C*H*W;
  if (idx >= total) return;
  int wc = idx % W, h = (idx/W) % H, c = (idx/(W*H)) % C, b = idx/(W*H*C);
  float sy = (h + 0.5f) * (16.0f / H) - 0.5f;
  float sx = (wc + 0.5f) * (16.0f / W) - 0.5f;
  sy = fminf(fmaxf(sy, 0.f), 15.f);
  sx = fminf(fmaxf(sx, 0.f), 15.f);
  int y0 = (int)sy, x0 = (int)sx;
  int y1 = min(y0+1, 15), x1 = min(x0+1, 15);
  float fy = sy - y0, fx = sx - x0;
  const float* mb = mseq + (size_t)b*256*C + c;
  float v00 = mb[(y0*16 + x0)*C], v01 = mb[(y0*16 + x1)*C];
  float v10 = mb[(y1*16 + x0)*C], v11 = mb[(y1*16 + x1)*C];
  float v = v00*(1-fy)*(1-fx) + v01*(1-fy)*fx + v10*fy*(1-fx) + v11*fy*fx;
  out[idx] = fmaxf(v + ident[idx], 0.f);
}

// ---------------------------------------------------------------------------
// CBAM
// ---------------------------------------------------------------------------
__global__ void k_chanstat(const float* __restrict__ x, float* __restrict__ avg,
                           float* __restrict__ mx, int C, int HW){
  __shared__ float s_sum[256], s_max[256];
  int bc = blockIdx.x, tid = threadIdx.x;
  const float* base = x + (size_t)bc*HW;
  float sum = 0.f, m = -3.4e38f;
  for (int i = tid; i < HW; i += 256) { float v = base[i]; sum += v; m = fmaxf(m, v); }
  s_sum[tid] = sum; s_max[tid] = m;
  __syncthreads();
  for (int st = 128; st > 0; st >>= 1) {
    if (tid < st) { s_sum[tid] += s_sum[tid+st]; s_max[tid] = fmaxf(s_max[tid], s_max[tid+st]); }
    __syncthreads();
  }
  if (tid == 0) { avg[bc] = s_sum[0]/(float)HW; mx[bc] = s_max[0]; }
}

__global__ void k_catt(const float* __restrict__ avg, const float* __restrict__ mx,
                       const float* __restrict__ w1, const float* __restrict__ b1,
                       const float* __restrict__ w2, const float* __restrict__ b2,
                       float* __restrict__ att, int C, int r){
  __shared__ float ha[64], hm[64];
  int b = blockIdx.x, tid = threadIdx.x;
  if (tid < r) {
    float sa = b1[tid], sm = b1[tid];
    for (int c = 0; c < C; ++c) {
      sa += w1[tid*C + c] * avg[b*C + c];
      sm += w1[tid*C + c] * mx[b*C + c];
    }
    ha[tid] = fmaxf(sa, 0.f);
    hm[tid] = fmaxf(sm, 0.f);
  }
  __syncthreads();
  for (int c = tid; c < C; c += blockDim.x) {
    float oa = b2[c], om = b2[c];
    for (int j = 0; j < r; ++j) { oa += w2[c*r + j]*ha[j]; om += w2[c*r + j]*hm[j]; }
    att[b*C + c] = sigmoidf_(oa + om);
  }
}

__global__ void k_scale_att(float* __restrict__ x, const float* __restrict__ att,
                            int C, int HW, int total){
  int idx = blockIdx.x*blockDim.x + threadIdx.x;
  if (idx >= total) return;
  int c = (idx/HW) % C, b = idx/(HW*C);
  x[idx] *= att[b*C + c];
}

__global__ void k_spatialpool(const float* __restrict__ x, float* __restrict__ cp,
                              int B, int C, int HW){
  int idx = blockIdx.x*blockDim.x + threadIdx.x;
  if (idx >= B*HW) return;
  int i = idx % HW, b = idx / HW;
  float m = -3.4e38f, s = 0.f;
  for (int c = 0; c < C; ++c) {
    float v = x[((size_t)(b*C + c))*HW + i];
    m = fmaxf(m, v); s += v;
  }
  cp[((size_t)(b*2 + 0))*HW + i] = m;
  cp[((size_t)(b*2 + 1))*HW + i] = s / (float)C;
}

__global__ void k_sigmoid_scale(float* __restrict__ x, const float* __restrict__ s,
                                int C, int HW, int total){
  int idx = blockIdx.x*blockDim.x + threadIdx.x;
  if (idx >= total) return;
  int i = idx % HW, b = idx/(HW*C);
  x[idx] *= sigmoidf_(s[(size_t)b*HW + i]);
}

// ---------------------------------------------------------------------------
// Decoder: transposed conv 2x2 stride 2 into concat buffer, skip copy
// ---------------------------------------------------------------------------
__global__ void k_upsample(const float* __restrict__ x, const float* __restrict__ up_w,
                           const float* __restrict__ up_b, float* __restrict__ cat,
                           int B, int Ci, int Co, int H, int W, int Ccat){
  int idx = blockIdx.x*blockDim.x + threadIdx.x;
  int H2 = 2*H, W2 = 2*W;
  int total = B*Co*H2*W2;
  if (idx >= total) return;
  int xo = idx % W2, yo = (idx/W2) % H2, o = (idx/(W2*H2)) % Co, b = idx/(W2*H2*Co);
  int h = yo >> 1, k = yo & 1, w = xo >> 1, l = xo & 1;
  float acc = up_b[o];
  for (int i = 0; i < Ci; ++i)
    acc += x[((size_t)(b*Ci + i))*H*W + h*W + w] * up_w[((size_t)(i*Co + o))*4 + k*2 + l];
  cat[((size_t)(b*Ccat + o))*H2*W2 + yo*W2 + xo] = acc;
}

__global__ void k_copy_skip(const float* __restrict__ skip, float* __restrict__ cat,
                            int B, int Co, int H2, int W2, int Ccat){
  int idx = blockIdx.x*blockDim.x + threadIdx.x;
  int total = B*Co*H2*W2;
  if (idx >= total) return;
  int i = idx % (H2*W2), c = (idx/(H2*W2)) % Co, b = idx/(H2*W2*Co);
  cat[((size_t)(b*Ccat + Co + c))*H2*W2 + i] = skip[idx];
}

// ---------------------------------------------------------------------------
// Seg head resize (hin,win) -> (512,512), and final average
// ---------------------------------------------------------------------------
__global__ void k_resize_out(const float* __restrict__ src, float* __restrict__ dst,
                             int B, int hin, int win, int S){
  int idx = blockIdx.x*blockDim.x + threadIdx.x;
  int total = B*S*S;
  if (idx >= total) return;
  int xo = idx % S, yo = (idx/S) % S, b = idx/(S*S);
  float sy = (yo + 0.5f) * ((float)hin / S) - 0.5f;
  float sx = (xo + 0.5f) * ((float)win / S) - 0.5f;
  sy = fminf(fmaxf(sy, 0.f), (float)(hin-1));
  sx = fminf(fmaxf(sx, 0.f), (float)(win-1));
  int y0 = (int)sy, x0 = (int)sx;
  int y1 = min(y0+1, hin-1), x1 = min(x0+1, win-1);
  float fy = sy - y0, fx = sx - x0;
  const float* sb = src + (size_t)b*hin*win;
  float v00 = sb[y0*win + x0], v01 = sb[y0*win + x1];
  float v10 = sb[y1*win + x0], v11 = sb[y1*win + x1];
  dst[idx] = v00*(1-fy)*(1-fx) + v01*(1-fy)*fx + v10*fy*(1-fx) + v11*fy*fx;
}

__global__ void k_final_avg(const float* __restrict__ o1, const float* __restrict__ o2,
                            const float* __restrict__ o3, float* __restrict__ fin, int n){
  int idx = blockIdx.x*blockDim.x + threadIdx.x;
  if (idx < n) fin[idx] = (o1[idx] + o2[idx] + o3[idx]) * (1.0f/3.0f);
}

// ===========================================================================
// Host orchestration
// ===========================================================================
namespace {

struct MambaPtrs {
  const float *in_w, *conv_w, *conv_b, *xproj_w, *dt_w, *dt_b, *A_log, *D, *out_w;
};
struct CbamPtrs {
  const float *w1, *b1, *w2, *b2, *sp_w, *sp_g, *sp_b;
};
struct Scratch {
  float *e1,*e2,*e3,*e4,*d3,*d2,*d1;
  float *cat,*conv,*ident,*seq,*xz,*xconv,*dbl,*dt,*y,*m;
  unsigned short *Abf, *Bbf;
  float *mean,*var,*avg,*mx,*att,*cp,*smap;
};

inline int gdiv(int a, int b){ return (a + b - 1)/b; }
inline int pad32(int n){ return (n + 31) & ~31; }
#define L256(kern, n, ...) kern<<<gdiv((n),256), 256, 0, stream>>>(__VA_ARGS__)

void run_mamba(hipStream_t stream, Scratch& S, const MambaPtrs& mp, int C){
  const int B = 4, L = 256, M = B*L;              // 1024
  const int di = 2*C, dr = (C + 15)/16, NC = dr + 128;
  const int NCp = pad32(NC);                       // 160 for all C here

  // xz = seq @ in_w.T   (M,C)x(C,2di)   -- N=2di is a multiple of 32
  L256(k_f32_to_bf16, M*C, S.seq, S.Abf, M*C);
  L256(k_stageB, C*(2*di), mp.in_w, S.Bbf, 2*di, C, 2*di);
  k_gemm_bf16_wmma<<<dim3((2*di)/32, M/16), 32, 0, stream>>>(S.Abf, S.Bbf, S.xz, M, 2*di, C, 2*di);

  // causal depthwise conv1d + silu
  L256(k_conv1d_silu, B*L*di, S.xz, mp.conv_w, mp.conv_b, S.xconv, B, L, di);

  // dbl = x @ xproj_w.T   (M,di)x(di,NC), staged N padded to NCp
  L256(k_f32_to_bf16, M*di, S.xconv, S.Abf, M*di);
  L256(k_stageB, di*NCp, mp.xproj_w, S.Bbf, NC, di, NCp);
  k_gemm_bf16_wmma<<<dim3(NCp/32, M/16), 32, 0, stream>>>(S.Abf, S.Bbf, S.dbl, M, NC, di, NCp);

  // dt = softplus(dt_proj)
  L256(k_dtproj, M*di, S.dbl, mp.dt_w, mp.dt_b, S.dt, M, di, dr, NC);

  // selective scan + gating
  L256(k_scan, B*di, S.dt, S.xconv, S.dbl, S.xz, mp.A_log, mp.D, S.y, B, di, dr, NC);

  // m = y @ out_w.T   (M,di)x(di,C)   -- C multiple of 32
  L256(k_f32_to_bf16, M*di, S.y, S.Abf, M*di);
  L256(k_stageB, di*C, mp.out_w, S.Bbf, C, di, C);
  k_gemm_bf16_wmma<<<dim3(C/32, M/16), 32, 0, stream>>>(S.Abf, S.Bbf, S.m, M, C, di, C);
}

void run_mcb(hipStream_t stream, Scratch& S, const float* x, int Cin, int Cout, int H, int W,
             const float* conv_w, const float* bn_g, const float* bn_b,
             const float* res_w, const float* res_g, const float* res_b,
             const MambaPtrs& mp, float* out){
  const int B = 4, HW = H*W;
  const int n = B*Cout*HW;
  // h = relu(bn(conv3x3(x)))
  L256(k_conv3x3, n, x, conv_w, S.conv, B, Cin, Cout, H, W);
  k_bn_stats<<<Cout, 256, 0, stream>>>(S.conv, S.mean, S.var, Cout, HW, B);
  L256(k_bn_apply, n, S.conv, S.conv, S.mean, S.var, bn_g, bn_b, Cout, HW, n, 1);
  // identity
  const float* ident = x;
  if (res_w) {
    L256(k_conv1x1, n, x, res_w, (const float*)nullptr, S.ident, B, Cin, Cout, H, W);
    k_bn_stats<<<Cout, 256, 0, stream>>>(S.ident, S.mean, S.var, Cout, HW, B);
    L256(k_bn_apply, n, S.ident, S.ident, S.mean, S.var, res_g, res_b, Cout, HW, n, 0);
    ident = S.ident;
  }
  // pool -> sequence
  L256(k_pool_seq, B*Cout*256, S.conv, S.seq, B, Cout, H, W);
  // mamba
  run_mamba(stream, S, mp, Cout);
  // upscale + residual + relu
  L256(k_resize_add_relu, n, S.m, ident, out, B, Cout, H, W);
}

void run_cbam(hipStream_t stream, Scratch& S, float* x, int C, int H, int W, const CbamPtrs& cp){
  const int B = 4, HW = H*W, n = B*C*HW, r = C/16;
  k_chanstat<<<B*C, 256, 0, stream>>>(x, S.avg, S.mx, C, HW);
  k_catt<<<B, 256, 0, stream>>>(S.avg, S.mx, cp.w1, cp.b1, cp.w2, cp.b2, S.att, C, r);
  L256(k_scale_att, n, x, S.att, C, HW, n);
  L256(k_spatialpool, B*HW, x, S.cp, B, C, HW);
  L256(k_conv7x7_2to1, B*HW, S.cp, cp.sp_w, S.smap, B, H, W);
  k_bn_stats<<<1, 256, 0, stream>>>(S.smap, S.mean, S.var, 1, HW, B);
  L256(k_bn_apply, B*HW, S.smap, S.smap, S.mean, S.var, cp.sp_g, cp.sp_b, 1, HW, B*HW, 0);
  L256(k_sigmoid_scale, n, x, S.smap, C, HW, n);
}

} // anonymous namespace

extern "C" void kernel_launch(void* const* d_in, const int* in_sizes, int n_in,
                              void* d_out, int out_size, void* d_ws, size_t ws_size,
                              hipStream_t stream) {
  auto IN = [&](int i) -> const float* { return (const float*)d_in[i]; };

  // ---- workspace bump allocator (floats) ----
  float* base = (float*)d_ws;
  size_t off = 0;
  auto alloc = [&](size_t nf) -> float* { float* p = base + off; off += (nf + 63) & ~(size_t)63; return p; };

  Scratch S;
  S.e1 = alloc(4*64*128*128);   S.e2 = alloc(4*128*64*64);
  S.e3 = alloc(4*320*32*32);    S.e4 = alloc(4*512*16*16);
  S.d3 = alloc(4*320*32*32);    S.d2 = alloc(4*128*64*64);
  S.d1 = alloc(4*64*128*128);
  S.cat   = alloc(4*128*128*128);      // largest concat input (dec1)
  S.conv  = alloc(4*64*128*128);
  S.ident = alloc(4*64*128*128);
  S.seq   = alloc(4*256*512);
  S.xz    = alloc(1024*2048);
  S.xconv = alloc(1024*1024);
  S.dbl   = alloc(1024*160);
  S.dt    = alloc(1024*1024);
  S.y     = alloc(1024*1024);
  S.m     = alloc(1024*512);
  S.Abf   = (unsigned short*)alloc(1024*1024/2 + 64);   // 1024x1024 ushorts
  S.Bbf   = (unsigned short*)alloc(1024*2048/2 + 64);   // up to 512x2048 ushorts
  S.mean  = alloc(1024); S.var = alloc(1024);
  S.avg   = alloc(4096); S.mx  = alloc(4096); S.att = alloc(4096);
  S.cp    = alloc(4*2*128*128);
  S.smap  = alloc(4*128*128);

  // ---- input index map (setup_inputs dict flattening, depth-first) ----
  // s1..s4 = 0..3; enc blocks (19 each): conv_w,bn_g,bn_b, mamba[9], cbam[7]
  // dec blocks (24 each): up_w,up_b, conv_w,bn_g,bn_b, mamba[9], res_w,res_g,res_b, cbam[7]
  auto mambaAt = [&](int i) {
    MambaPtrs m = { IN(i), IN(i+1), IN(i+2), IN(i+3), IN(i+4), IN(i+5), IN(i+6), IN(i+7), IN(i+8) };
    return m;
  };
  auto cbamAt = [&](int i) {
    CbamPtrs c = { IN(i), IN(i+1), IN(i+2), IN(i+3), IN(i+4), IN(i+5), IN(i+6) };
    return c;
  };

  const int ENC[4] = {4, 23, 42, 61};
  const int DEC3 = 80, DEC2 = 104, DEC1 = 128;
  const int SEG3 = 152, SEG2 = 154, SEG1 = 156;

  const float* s_in[4] = { IN(0), IN(1), IN(2), IN(3) };
  const int   encC[4]  = { 64, 128, 320, 512 };
  const int   encH[4]  = { 128, 64, 32, 16 };
  float* encOut[4] = { S.e1, S.e2, S.e3, S.e4 };

  // ---- encoders ----
  for (int i = 0; i < 4; ++i) {
    int b = ENC[i];
    MambaPtrs mp = mambaAt(b+3);
    run_mcb(stream, S, s_in[i], encC[i], encC[i], encH[i], encH[i],
            IN(b), IN(b+1), IN(b+2), nullptr, nullptr, nullptr, mp, encOut[i]);
    run_cbam(stream, S, encOut[i], encC[i], encH[i], encH[i], cbamAt(b+12));
  }

  // ---- decoders ----
  struct DecCfg { int base; const float* x; const float* skip; int Ci, Co, H; float* out; };
  DecCfg dcs[3] = {
    { DEC3, S.e4, S.e3, 512, 320, 16, S.d3 },
    { DEC2, S.d3, S.e2, 320, 128, 32, S.d2 },
    { DEC1, S.d2, S.e1, 128,  64, 64, S.d1 },
  };
  for (int i = 0; i < 3; ++i) {
    DecCfg& D = dcs[i];
    int b = D.base;
    int H2 = 2*D.H, W2 = 2*D.H, Ccat = 2*D.Co;
    L256(k_upsample, 4*D.Co*H2*W2, D.x, IN(b), IN(b+1), S.cat, 4, D.Ci, D.Co, D.H, D.H, Ccat);
    L256(k_copy_skip, 4*D.Co*H2*W2, D.skip, S.cat, 4, D.Co, H2, W2, Ccat);
    MambaPtrs mp = mambaAt(b+5);
    run_mcb(stream, S, S.cat, Ccat, D.Co, H2, W2,
            IN(b+2), IN(b+3), IN(b+4), IN(b+14), IN(b+15), IN(b+16), mp, D.out);
    run_cbam(stream, S, D.out, D.Co, H2, W2, cbamAt(b+17));
  }

  // ---- seg heads ----
  const int Sz = 512, BSS = 4*Sz*Sz;
  float* o1 = (float*)d_out;
  float* o2 = o1 + BSS;
  float* o3 = o2 + BSS;
  float* fin = o3 + BSS;

  // o3 from d3 (320ch, 32x32)
  L256(k_conv1x1, 4*32*32, S.d3, IN(SEG3), IN(SEG3+1), S.smap, 4, 320, 1, 32, 32);
  L256(k_resize_out, BSS, S.smap, o3, 4, 32, 32, Sz);
  // o2 from d2 (128ch, 64x64)
  L256(k_conv1x1, 4*64*64, S.d2, IN(SEG2), IN(SEG2+1), S.smap, 4, 128, 1, 64, 64);
  L256(k_resize_out, BSS, S.smap, o2, 4, 64, 64, Sz);
  // o1 from d1 (64ch, 128x128)
  L256(k_conv1x1, 4*128*128, S.d1, IN(SEG1), IN(SEG1+1), S.smap, 4, 64, 1, 128, 128);
  L256(k_resize_out, BSS, S.smap, o1, 4, 128, 128, Sz);

  L256(k_final_avg, BSS, o1, o2, o3, fin, BSS);

  (void)in_sizes; (void)n_in; (void)out_size; (void)ws_size;
}
